// OpenCLIPAttention_85718957294009
// MI455X (gfx1250) — matrix operand order repaired and checked by reading, not compiled, not run on hardware
//
#include <hip/hip_runtime.h>

typedef __bf16 bf16;
typedef __attribute__((ext_vector_type(16))) __bf16 v16bf;
typedef __attribute__((ext_vector_type(8)))  float  v8f;

#define D_MODEL 1024
#define NH      16
#define DH      64
#define BATCH   8
#define SEQ     1024
#define ROWS    (BATCH * SEQ)   /* 8192 */
#define QKV_OUT (3 * D_MODEL)   /* 3072 */

union FragU { v16bf v; uint4 u[2]; };

// ---- WMMA wrapper: D = A(16x32 bf16) x B(32x16 bf16) + C(16x16 f32) ----
__device__ __forceinline__ v8f wmma_bf16(v16bf a, v16bf b, v8f c) {
  return __builtin_amdgcn_wmma_f32_16x16x32_bf16(
      /*neg_a=*/false, a, /*neg_b=*/false, b,
      /*c_mod=*/(short)0, c, /*reuse_a=*/false, /*reuse_b=*/false);
}

// A-frag (16x32, MxK) from row-major A[lda]: lane m = lane&15, half = lane>>4
// elements 0..7 -> K = k0 + half*8 + e ; elements 8..15 -> K = k0+16+half*8+e
__device__ __forceinline__ v16bf load_a_frag_g(const bf16* __restrict__ A,
                                               int lda, int m0, int k0) {
  const int lane = threadIdx.x & 31;
  const int m    = m0 + (lane & 15);
  const int half = lane >> 4;
  const bf16* p  = A + (size_t)m * lda + k0 + half * 8;
  FragU f;
  f.u[0] = *(const uint4*)p;
  f.u[1] = *(const uint4*)(p + 16);
  return f.v;
}

// B-frag (32x16, KxN) where B[k][n] = W[n0+n][k0+k], W row-major [*, ldw]:
// lane n = lane&15, elements 0..7 -> K = k0 + (lane>>4)*8 + e ; elements 8..15 -> K = k0+16+(lane>>4)*8+e
__device__ __forceinline__ v16bf load_b_frag_g(const bf16* __restrict__ W,
                                               int ldw, int n0, int k0) {
  const int lane = threadIdx.x & 31;
  const int n    = n0 + (lane & 15);
  const int half = lane >> 4;
  const bf16* p  = W + (size_t)n * ldw + k0 + half * 8;
  FragU f;
  f.u[0] = *(const uint4*)p;
  f.u[1] = *(const uint4*)(p + 16);
  return f.v;
}

// ---- f32 -> bf16 conversion ----
__global__ void cvt_f32_bf16(const float* __restrict__ in, bf16* __restrict__ out, int n) {
  int i = blockIdx.x * blockDim.x + threadIdx.x;
  const int stride = gridDim.x * blockDim.x;
  for (; i < n; i += stride) out[i] = (bf16)in[i];
}

// ---- Fused QKV GEMM: [8192,1024] x [3072,1024]^T + bias, scatter to Q,K,V^T ----
// Q,K: [B*H, N, Dh] bf16 (Q pre-scaled by 1/sqrt(Dh)); V^T: [B*H, Dh, N] bf16
__global__ void __launch_bounds__(256) qkv_gemm(
    const bf16* __restrict__ X, const bf16* __restrict__ W,
    const float* __restrict__ bias,
    bf16* __restrict__ qws, bf16* __restrict__ kws, bf16* __restrict__ vtws) {
  const int wave = threadIdx.x >> 5;
  const int lane = threadIdx.x & 31;
  const int tile = blockIdx.x * 8 + wave;
  const int TN   = QKV_OUT / 64;          // 48 n-tiles
  const int m0   = (tile / TN) * 32;
  const int n0   = (tile % TN) * 64;

  v8f acc[2][4];
#pragma unroll
  for (int i = 0; i < 2; ++i)
#pragma unroll
    for (int j = 0; j < 4; ++j) acc[i][j] = (v8f){};

  for (int k0 = 0; k0 < D_MODEL; k0 += 32) {
    v16bf a0 = load_a_frag_g(X, D_MODEL, m0, k0);
    v16bf a1 = load_a_frag_g(X, D_MODEL, m0 + 16, k0);
#pragma unroll
    for (int j = 0; j < 4; ++j) {
      v16bf b = load_b_frag_g(W, D_MODEL, n0 + 16 * j, k0);
      acc[0][j] = wmma_bf16(a0, b, acc[0][j]);
      acc[1][j] = wmma_bf16(a1, b, acc[1][j]);
    }
  }

  const int half = lane >> 4, ln = lane & 15;
#pragma unroll
  for (int i = 0; i < 2; ++i)
#pragma unroll
    for (int j = 0; j < 4; ++j) {
      const int   c     = n0 + 16 * j + ln;
      const int   which = c >> 10;        // 0=q 1=k 2=v
      const int   h     = (c >> 6) & 15;
      const int   d     = c & 63;
      const float bval  = bias[c];
#pragma unroll
      for (int e = 0; e < 8; ++e) {
        const int   r  = m0 + 16 * i + e + 8 * half;
        const int   b_ = r >> 10, n = r & 1023;
        const int   bh = (b_ << 4) + h;
        const float v  = acc[i][j][e] + bval;
        if (which == 0)
          qws[((size_t)bh * SEQ + n) * DH + d] = (bf16)(v * 0.125f);
        else if (which == 1)
          kws[((size_t)bh * SEQ + n) * DH + d] = (bf16)v;
        else
          vtws[((size_t)bh * DH + d) * SEQ + n] = (bf16)v;
      }
    }
}

// A-frag from per-wave LDS f32 tile P[16][32] (row-major), converting to bf16
__device__ __forceinline__ v16bf a_frag_from_lds(const float* __restrict__ P) {
  const int lane = threadIdx.x & 31;
  const int m = lane & 15, half = lane >> 4;
  const float* row = P + m * 32;
  v16bf r;
#pragma unroll
  for (int e = 0; e < 8; ++e) r[e] = (bf16)row[half * 8 + e];
#pragma unroll
  for (int e = 0; e < 8; ++e) r[8 + e] = (bf16)row[16 + half * 8 + e];
  return r;
}

// ---- Flash attention: one wave handles 16 query rows of one (b,h) ----
__global__ void __launch_bounds__(256) attn_flash(
    const bf16* __restrict__ Q, const bf16* __restrict__ K,
    const bf16* __restrict__ VT, bf16* __restrict__ AO) {
  __shared__ float Plds[8][16 * 32];      // 16 KB: per-wave P tile
  const int wave  = threadIdx.x >> 5;
  const int lane  = threadIdx.x & 31;
  const int gw    = blockIdx.x * 8 + wave;      // 0..8191
  const int qtile = gw & 63;
  const int bh    = gw >> 6;                    // 0..127
  const int b     = bh >> 4, h = bh & 15;
  const int half  = lane >> 4, ln = lane & 15;

  const bf16* Qm = Q  + (size_t)bh * SEQ * DH;  // [1024][64]
  const bf16* Km = K  + (size_t)bh * SEQ * DH;  // [1024][64]
  const bf16* Vt = VT + (size_t)bh * DH * SEQ;  // [64][1024]

  const int q0 = qtile * 16;
  const v16bf aq0 = load_a_frag_g(Qm, DH, q0, 0);   // d = 0..31
  const v16bf aq1 = load_a_frag_g(Qm, DH, q0, 32);  // d = 32..63

  v8f accO[4];
#pragma unroll
  for (int t = 0; t < 4; ++t) accO[t] = (v8f){};
  float mrow[8], lrow[8];
#pragma unroll
  for (int i = 0; i < 8; ++i) { mrow[i] = -1e30f; lrow[i] = 0.0f; }

  float* P = &Plds[wave][0];

  for (int c0 = 0; c0 < SEQ; c0 += 32) {
    // S = q . k^T for 32 keys (scale already folded into Q)
    v8f s0 = (v8f){}, s1 = (v8f){};
    s0 = wmma_bf16(aq0, load_b_frag_g(Km, DH, c0,       0),  s0);
    s0 = wmma_bf16(aq1, load_b_frag_g(Km, DH, c0,       32), s0);
    s1 = wmma_bf16(aq0, load_b_frag_g(Km, DH, c0 + 16,  0),  s1);
    s1 = wmma_bf16(aq1, load_b_frag_g(Km, DH, c0 + 16,  32), s1);

    // online softmax; row of element i is (i + 8*half), reductions stay in-half
    float p0[8], p1[8];
#pragma unroll
    for (int i = 0; i < 8; ++i) {
      float lm = fmaxf(s0[i], s1[i]);
#pragma unroll
      for (int off = 1; off < 16; off <<= 1) lm = fmaxf(lm, __shfl_xor(lm, off, 32));
      const float mnew  = fmaxf(mrow[i], lm);
      const float alpha = __expf(mrow[i] - mnew);
      mrow[i] = mnew;
      const float e0 = __expf(s0[i] - mnew);
      const float e1 = __expf(s1[i] - mnew);
      p0[i] = e0; p1[i] = e1;
      float rs = e0 + e1;
#pragma unroll
      for (int off = 1; off < 16; off <<= 1) rs += __shfl_xor(rs, off, 32);
      lrow[i] = lrow[i] * alpha + rs;
#pragma unroll
      for (int t = 0; t < 4; ++t) accO[t][i] *= alpha;
    }

    // C-layout -> A-frag layout via LDS (per-wave private tile)
#pragma unroll
    for (int i = 0; i < 8; ++i) {
      const int row = i + 8 * half;
      P[row * 32 + ln]      = p0[i];
      P[row * 32 + ln + 16] = p1[i];
    }
    __syncthreads();
    const v16bf ap = a_frag_from_lds(P);

    // accO += P . V  (V^T rows are contiguous in key index)
#pragma unroll
    for (int t = 0; t < 4; ++t)
      accO[t] = wmma_bf16(ap, load_b_frag_g(Vt, SEQ, 16 * t, c0), accO[t]);
    __syncthreads();
  }

  // normalize and write to [B, N, H*Dh] bf16
#pragma unroll
  for (int t = 0; t < 4; ++t) {
    const int d = 16 * t + ln;
#pragma unroll
    for (int i = 0; i < 8; ++i) {
      const int   n = q0 + i + 8 * half;
      const float v = accO[t][i] / lrow[i];
      AO[((size_t)b * SEQ + n) * D_MODEL + h * DH + d] = (bf16)v;
    }
  }
}

// ---- Output projection: [8192,1024] x [1024,1024]^T + bias -> f32 ----
__global__ void __launch_bounds__(256) proj_gemm(
    const bf16* __restrict__ X, const bf16* __restrict__ W,
    const float* __restrict__ bias, float* __restrict__ out) {
  const int wave = threadIdx.x >> 5;
  const int lane = threadIdx.x & 31;
  const int tile = blockIdx.x * 8 + wave;
  const int TN   = D_MODEL / 64;          // 16 n-tiles
  const int m0   = (tile / TN) * 32;
  const int n0   = (tile % TN) * 64;

  v8f acc[2][4];
#pragma unroll
  for (int i = 0; i < 2; ++i)
#pragma unroll
    for (int j = 0; j < 4; ++j) acc[i][j] = (v8f){};

  for (int k0 = 0; k0 < D_MODEL; k0 += 32) {
    v16bf a0 = load_a_frag_g(X, D_MODEL, m0, k0);
    v16bf a1 = load_a_frag_g(X, D_MODEL, m0 + 16, k0);
#pragma unroll
    for (int j = 0; j < 4; ++j) {
      v16bf b = load_b_frag_g(W, D_MODEL, n0 + 16 * j, k0);
      acc[0][j] = wmma_bf16(a0, b, acc[0][j]);
      acc[1][j] = wmma_bf16(a1, b, acc[1][j]);
    }
  }

  const int half = lane >> 4, ln = lane & 15;
#pragma unroll
  for (int i = 0; i < 2; ++i)
#pragma unroll
    for (int j = 0; j < 4; ++j) {
      const int   c    = n0 + 16 * j + ln;
      const float bval = bias[c];
#pragma unroll
      for (int e = 0; e < 8; ++e) {
        const int r = m0 + 16 * i + e + 8 * half;
        out[(size_t)r * D_MODEL + c] = acc[i][j][e] + bval;
      }
    }
}

extern "C" void kernel_launch(void* const* d_in, const int* in_sizes, int n_in,
                              void* d_out, int out_size, void* d_ws, size_t ws_size,
                              hipStream_t stream) {
  const float* x      = (const float*)d_in[0];
  const float* qkv_w  = (const float*)d_in[1];
  const float* qkv_b  = (const float*)d_in[2];
  const float* proj_w = (const float*)d_in[3];
  const float* proj_b = (const float*)d_in[4];
  float*       out    = (float*)d_out;

  const size_t MB = (size_t)1 << 20;
  char* ws = (char*)d_ws;
  bf16* xb  = (bf16*)(ws + 0 * MB);    // 8M elems  (16 MB)
  bf16* wq  = (bf16*)(ws + 16 * MB);   // 3M elems  ( 6 MB)
  bf16* wp  = (bf16*)(ws + 22 * MB);   // 1M elems  ( 2 MB)
  bf16* qws = (bf16*)(ws + 24 * MB);   // 8M elems  (16 MB)
  bf16* kws = (bf16*)(ws + 40 * MB);   // 8M elems  (16 MB)
  bf16* vt  = (bf16*)(ws + 56 * MB);   // 8M elems  (16 MB)
  bf16* ao  = (bf16*)(ws + 72 * MB);   // 8M elems  (16 MB)

  cvt_f32_bf16<<<4096, 256, 0, stream>>>(x,      xb, ROWS * D_MODEL);
  cvt_f32_bf16<<<2048, 256, 0, stream>>>(qkv_w,  wq, QKV_OUT * D_MODEL);
  cvt_f32_bf16<<<1024, 256, 0, stream>>>(proj_w, wp, D_MODEL * D_MODEL);

  // (8192/32) * (3072/64) = 12288 wave tiles / 8 waves per block
  qkv_gemm<<<1536, 256, 0, stream>>>(xb, wq, qkv_b, qws, kws, vt);

  // 8192 waves (128 bh * 64 q-tiles) / 8 per block
  attn_flash<<<1024, 256, 0, stream>>>(qws, kws, vt, ao);

  // (8192/32) * (1024/64) = 4096 wave tiles / 8 waves per block
  proj_gemm<<<512, 256, 0, stream>>>(ao, wp, proj_b, out);
}
